// WaveNet_18519898980949
// MI455X (gfx1250) — matrix operand order repaired and checked by reading, not compiled, not run on hardware
//
#include <hip/hip_runtime.h>
#include <hip/hip_bf16.h>
#include <math.h>

typedef __attribute__((ext_vector_type(16))) _Float16 v16h;
typedef __attribute__((ext_vector_type(8)))  _Float16 v8h;
typedef __attribute__((ext_vector_type(8)))  float    v8f;

#define T_LEN   4096
#define BATCH   8
#define CLASSES 256
#define NSUB    4              // 4 x 16 = 64 time steps per block tile
#define LDS_STRIDE 40          // 32 k-rows + 8 pad halfs -> conflict-free b128 reads

// ---------------------------------------------------------------------------
// Pack conv weights (f32, [O][I][kt]) into WMMA A-fragment order (f16):
//   out[((otile*nChunk + chunk)*32 + lane)*16 + e]
// so the GEMM kernel loads each A fragment as two coalesced b128 loads.
// K ordering: k = tap*Cin + i.
// ---------------------------------------------------------------------------
__global__ void pack_w_kernel(const float* __restrict__ W,
                              _Float16* __restrict__ out,
                              int Cin, int ktaps, int oTiles)
{
    int tid  = blockIdx.x * blockDim.x + threadIdx.x;
    int lane = tid & 31;
    int rest = tid >> 5;
    int nChunk = (Cin * ktaps) >> 5;
    int chunk = rest % nChunk;
    int otile = rest / nChunk;
    if (otile >= oTiles) return;

    int laneM = lane & 15;
    int kloA  = (lane < 16) ? 0 : 8;
    int kk    = chunk * 32;

    v8h w0, w1;
    #pragma unroll
    for (int e = 0; e < 16; ++e) {
        int kloc = kloA + e + ((e & 8) ? 8 : 0);        // ISA 16-bit A K map
        int k    = kk + kloc;
        int tap  = (k >= Cin) ? 1 : 0;                  // ktaps <= 2
        int i    = k - tap * Cin;
        _Float16 h = (_Float16)W[((size_t)(otile * 16 + laneM) * Cin + i) * ktaps + tap];
        if (e < 8) w0[e] = h; else w1[e - 8] = h;
    }
    v8h* dst = (v8h*)(out + ((size_t)(otile * nChunk + chunk) * 32 + lane) * 16);
    dst[0] = w0;
    dst[1] = w1;
}

// ---------------------------------------------------------------------------
// Causal conv (k = 1 or 2) as LDS-staged, double-buffered WMMA GEMM.
//   y[b,o,t] = sum_k Wp[o,k] * x[b,i(k),t + dil*(tap(k)-(kt-1))] + bias[o]
// Workgroup = 8 waves = one (b, 64-time) window over ALL Cout (OPW o-tiles
// per wave; Cout = OPW*128). Each 32-K activation chunk is staged once in
// LDS (transposed [t][k], conflict-free) and shared by all 8 waves, with a
// software pipeline overlapping chunk c+1 global loads with chunk c WMMAs.
// ---------------------------------------------------------------------------
template<int OPW>
__global__ void conv_wmma_lds_kernel(const _Float16* __restrict__ xIn,
                                     const _Float16* __restrict__ Wp,
                                     const float*    __restrict__ bias,
                                     _Float16*       __restrict__ yOut,
                                     int Cin, int dil, int ktaps,
                                     int act, int resAdd)
{
    constexpr int Cout = OPW * 128;
    __shared__ __align__(16) _Float16 ldsB[2][64 * LDS_STRIDE];

    const int tid    = threadIdx.x;
    const int lane   = tid & 31;
    const int waveId = tid >> 5;
    const int b      = blockIdx.x >> 6;            // T_LEN/64 = 64 tiles
    const int t0     = (blockIdx.x & 63) * 64;

    const int laneM = lane & 15;                   // A row M / B col N
    const int kloB  = (lane < 16) ? 0 : 8;         // ISA 16-bit B layout
    const int K2    = Cin * ktaps;
    const int nChunk = K2 >> 5;

    // cooperative-fill mapping: 256 threads cover 32 rows x 64 cols, 8 cols each
    const int fi = tid >> 3;                       // k-row 0..31
    const int fc = (tid & 7) * 8;                  // col group 0,8,..,56

    v8f zero = {};
    v8f acc[OPW][NSUB];
    #pragma unroll
    for (int j = 0; j < OPW; ++j)
        #pragma unroll
        for (int s = 0; s < NSUB; ++s) acc[j][s] = zero;

    auto loadChunk = [&](int c) -> v8h {
        const int kk    = c * 32;
        const int tap   = (kk >= Cin) ? 1 : 0;     // ktaps <= 2
        const int iBase = kk - tap * Cin;
        const int toff  = (ktaps == 2) ? dil * (tap - 1) : 0;
        const int gt    = t0 + toff + fc;
        const _Float16* src = xIn + ((size_t)b * Cin + iBase + fi) * T_LEN;
        v8h v;
        if (gt >= 0 && (gt & 7) == 0) {
            v = *(const v8h*)(src + gt);           // aligned 16B global load
        } else {
            #pragma unroll
            for (int e = 0; e < 8; ++e) {
                int t = gt + e;
                v[e] = (t >= 0) ? src[t] : (_Float16)0.0f;
            }
        }
        return v;
    };

    v8h stage = loadChunk(0);
    for (int c = 0; c < nChunk; ++c) {
        // ---- commit staged chunk c into LDS buffer c&1 (transposed) ----
        _Float16* lb = &ldsB[c & 1][0];
        #pragma unroll
        for (int e = 0; e < 8; ++e)
            lb[(fc + e) * LDS_STRIDE + fi] = stage[e];
        __syncthreads();

        // ---- issue global loads for chunk c+1 (overlap with WMMAs) ----
        if (c + 1 < nChunk) stage = loadChunk(c + 1);

        // ---- A fragments: two coalesced b128 loads from packed weights ----
        v16h afrag[OPW];
        #pragma unroll
        for (int j = 0; j < OPW; ++j) {
            const v8h* wp = (const v8h*)(Wp +
                ((size_t)((waveId * OPW + j) * nChunk + c) * 32 + lane) * 16);
            v8h w0 = wp[0];
            v8h w1 = wp[1];
            #pragma unroll
            for (int e = 0; e < 8; ++e) { afrag[j][e] = w0[e]; afrag[j][e + 8] = w1[e]; }
        }

        // ---- B fragments from LDS (2x ds_load_b128 each) + WMMA ----
        #pragma unroll
        for (int s = 0; s < NSUB; ++s) {
            const int tcol = s * 16 + laneM;
            const v8h* lp = (const v8h*)&lb[tcol * LDS_STRIDE + kloB];
            v8h lo = lp[0];
            v8h hi = lp[2];
            v16h bf;
            #pragma unroll
            for (int e = 0; e < 8; ++e) { bf[e] = lo[e]; bf[e + 8] = hi[e]; }
            #pragma unroll
            for (int j = 0; j < OPW; ++j)
                acc[j][s] = __builtin_amdgcn_wmma_f32_16x16x32_f16(
                    false, afrag[j], false, bf, (short)0, acc[j][s],
                    false, false);
        }
    }

    // ---- epilogue: bias (+residual) (+relu), store f16 ----
    const int rowBase = (lane < 16) ? 0 : 8;       // ISA f32 C/D layout
    #pragma unroll
    for (int j = 0; j < OPW; ++j) {
        const int o0 = (waveId * OPW + j) * 16;
        #pragma unroll
        for (int s = 0; s < NSUB; ++s) {
            const int t = t0 + s * 16 + laneM;
            #pragma unroll
            for (int r = 0; r < 8; ++r) {
                const int o = o0 + rowBase + r;
                size_t idx = ((size_t)b * Cout + o) * T_LEN + t;
                float v = acc[j][s][r] + bias[o];
                if (resAdd) v += (float)yOut[idx];
                if (act == 1) v = v > 0.0f ? v : 0.0f;
                yOut[idx] = (_Float16)v;
            }
        }
    }
}

// ---------------------------------------------------------------------------
// mu-law companding of the raw input, f32 -> f16
// ---------------------------------------------------------------------------
__global__ void mulaw_kernel(const float* __restrict__ x,
                             _Float16* __restrict__ y, float invLog)
{
    int i = blockIdx.x * blockDim.x + threadIdx.x;   // BATCH*T_LEN
    float v = x[i];
    v = fminf(1.0f, fmaxf(-1.0f, v));
    float s = (v >= 0.0f) ? 1.0f : -1.0f;
    y[i] = (_Float16)(s * log1pf(256.0f * fabsf(v)) * invLog);
}

// ---------------------------------------------------------------------------
// 1 -> 64 causal conv (K too small for WMMA)
// ---------------------------------------------------------------------------
__global__ void conv_in_kernel(const _Float16* __restrict__ x,
                               const float* __restrict__ W,
                               const float* __restrict__ bias,
                               _Float16* __restrict__ y)
{
    int idx = blockIdx.x * blockDim.x + threadIdx.x;  // B*64*T
    int t = idx & (T_LEN - 1);
    int o = (idx >> 12) & 63;
    int b = idx >> 18;
    float xm1 = (t > 0) ? (float)x[(size_t)b * T_LEN + t - 1] : 0.0f;
    float x0  = (float)x[(size_t)b * T_LEN + t];
    float v = W[o * 2] * xm1 + W[o * 2 + 1] * x0 + bias[o];
    y[((size_t)b * 64 + o) * T_LEN + t] = (_Float16)v;
}

// ---------------------------------------------------------------------------
// z = tanh(f) * sigmoid(g)
// ---------------------------------------------------------------------------
__global__ void gate_kernel(const _Float16* __restrict__ f,
                            const _Float16* __restrict__ g,
                            _Float16* __restrict__ z)
{
    size_t i = (size_t)blockIdx.x * blockDim.x + threadIdx.x;
    float tf = tanhf((float)f[i]);
    float sg = 1.0f / (1.0f + __expf(-(float)g[i]));
    z[i] = (_Float16)(tf * sg);
}

// ---------------------------------------------------------------------------
// skip accumulation (f32) — init flag zeroes first use (ws is poisoned)
// ---------------------------------------------------------------------------
__global__ void skip_add_kernel(float* __restrict__ skip,
                                const _Float16* __restrict__ h, int init)
{
    size_t i = (size_t)blockIdx.x * blockDim.x + threadIdx.x;
    float prev = init ? 0.0f : skip[i];
    skip[i] = prev + (float)h[i];
}

// ---------------------------------------------------------------------------
// relu(skip f32) -> f16 staging buffer
// ---------------------------------------------------------------------------
__global__ void relu_f32_to_f16_kernel(const float* __restrict__ s,
                                       _Float16* __restrict__ r)
{
    size_t i = (size_t)blockIdx.x * blockDim.x + threadIdx.x;
    r[i] = (_Float16)fmaxf(s[i], 0.0f);
}

// ---------------------------------------------------------------------------
// softmax over 256 channels, one wave32 per (b,t) column, f16 -> f32 out
// ---------------------------------------------------------------------------
__global__ void softmax_kernel(const _Float16* __restrict__ logits,
                               float* __restrict__ out)
{
    int wave = (blockIdx.x * blockDim.x + threadIdx.x) >> 5;
    int lane = threadIdx.x & 31;
    int b = wave >> 12;            // 4096 columns per batch
    int t = wave & (T_LEN - 1);

    float v[8];
    float mx = -3.0e38f;
    #pragma unroll
    for (int j = 0; j < 8; ++j) {
        int c = lane + 32 * j;
        v[j] = (float)logits[((size_t)b * CLASSES + c) * T_LEN + t];
        mx = fmaxf(mx, v[j]);
    }
    #pragma unroll
    for (int off = 16; off > 0; off >>= 1)
        mx = fmaxf(mx, __shfl_xor(mx, off, 32));
    float sum = 0.0f;
    #pragma unroll
    for (int j = 0; j < 8; ++j) { v[j] = __expf(v[j] - mx); sum += v[j]; }
    #pragma unroll
    for (int off = 16; off > 0; off >>= 1)
        sum += __shfl_xor(sum, off, 32);
    float inv = 1.0f / sum;
    #pragma unroll
    for (int j = 0; j < 8; ++j) {
        int c = lane + 32 * j;
        out[((size_t)b * CLASSES + c) * T_LEN + t] = v[j] * inv;
    }
}

// ---------------------------------------------------------------------------
extern "C" void kernel_launch(void* const* d_in, const int* in_sizes, int n_in,
                              void* d_out, int out_size, void* d_ws, size_t ws_size,
                              hipStream_t stream)
{
    (void)in_sizes; (void)n_in; (void)out_size; (void)ws_size;

    struct ConvP { const float* W; const float* b; };
    const float* x_raw = (const float*)d_in[0];
    int p = 1;
    auto nxt = [&]() { return (const float*)d_in[p++]; };

    ConvP sconv[2][3], fprm[2][2][7], gprm[2][2][7], scprm[2][2];
    for (int s = 0; s < 2; ++s) {
        for (int c = 0; c < 3; ++c) { sconv[s][c].W = nxt(); sconv[s][c].b = nxt(); }
        for (int blk = 0; blk < 2; ++blk) {
            for (int l = 0; l < 7; ++l) { fprm[s][blk][l].W = nxt(); fprm[s][blk][l].b = nxt(); }
            for (int l = 0; l < 7; ++l) { gprm[s][blk][l].W = nxt(); gprm[s][blk][l].b = nxt(); }
            scprm[s][blk].W = nxt(); scprm[s][blk].b = nxt();
        }
    }
    ConvP c1; c1.W = nxt(); c1.b = nxt();
    ConvP c2; c2.W = nxt(); c2.b = nxt();

    // ---- workspace carve-out (~118 MB) ----
    const size_t ACT = (size_t)BATCH * CLASSES * T_LEN;   // 8M elements
    size_t off = 0;
    auto alloc = [&](size_t bytes) {
        void* ptr = (char*)d_ws + off;
        off += (bytes + 255) & ~(size_t)255;
        return ptr;
    };
    _Float16* xin  = (_Float16*)alloc((size_t)BATCH * T_LEN * 2);
    _Float16* hbuf = (_Float16*)alloc(ACT * 2);
    _Float16* abuf = (_Float16*)alloc(ACT * 2);
    _Float16* bbuf = (_Float16*)alloc(ACT * 2);
    _Float16* zf   = (_Float16*)alloc(ACT * 2);
    _Float16* zg   = (_Float16*)alloc(ACT * 2);
    float*    skip = (float*)   alloc(ACT * 4);

    // ---- pre-pack all WMMA-layer weights into fragment order (f16) ----
    auto packLayer = [&](const ConvP& cp, int Cin, int kt, int Cout) -> _Float16* {
        _Float16* dst = (_Float16*)alloc((size_t)Cout * Cin * kt * 2);
        int oTiles = Cout / 16;
        int nChunk = (Cin * kt) >> 5;
        int threads = oTiles * nChunk * 32;
        pack_w_kernel<<<(threads + 255) / 256, 256, 0, stream>>>(
            cp.W, dst, Cin, kt, oTiles);
        return dst;
    };

    static const int fCin[7]  = {256, 128, 128, 128, 128, 128, 128};
    static const int fCout[7] = {128, 128, 128, 128, 128, 128, 256};

    _Float16 *pk_sconv[2][2], *pk_f[2][2][7], *pk_g[2][2][7], *pk_sc[2][2];
    for (int s = 0; s < 2; ++s) {
        pk_sconv[s][0] = packLayer(sconv[s][1], 64, 2, 128);
        pk_sconv[s][1] = packLayer(sconv[s][2], 128, 2, 256);
        for (int blk = 0; blk < 2; ++blk) {
            for (int l = 0; l < 7; ++l) {
                pk_f[s][blk][l] = packLayer(fprm[s][blk][l], fCin[l], 2, fCout[l]);
                pk_g[s][blk][l] = packLayer(gprm[s][blk][l], fCin[l], 2, fCout[l]);
            }
            pk_sc[s][blk] = packLayer(scprm[s][blk], 256, 1, 256);
        }
    }
    _Float16* pk_c1 = packLayer(c1, 256, 1, 256);
    _Float16* pk_c2 = packLayer(c2, 256, 1, 256);

    auto convL = [&](const _Float16* xi, const _Float16* pk, const float* bias,
                     _Float16* yo, int Cin, int Cout, int dil, int kt,
                     int act, int res) {
        const int blocks = BATCH * (T_LEN / 64);          // 512
        if (Cout == 256)
            conv_wmma_lds_kernel<2><<<blocks, 256, 0, stream>>>(
                xi, pk, bias, yo, Cin, dil, kt, act, res);
        else
            conv_wmma_lds_kernel<1><<<blocks, 256, 0, stream>>>(
                xi, pk, bias, yo, Cin, dil, kt, act, res);
    };

    // dilated stack: 256->128 d1, 5x 128->128 d2..32, 128->256 d64 -> out
    auto dstack = [&](const _Float16* src, const ConvP* L, _Float16* const* pk,
                      _Float16* out) {
        convL(src, pk[0], L[0].b, abuf, 256, 128, 1, 2, 0, 0);
        const _Float16* cur = abuf;
        int dil = 2;
        for (int l = 1; l < 6; ++l) {
            _Float16* dst = (cur == abuf) ? bbuf : abuf;
            convL(cur, pk[l], L[l].b, dst, 128, 128, dil, 2, 0, 0);
            cur = dst;
            dil <<= 1;
        }
        convL(cur, pk[6], L[6].b, out, 128, 256, 64, 2, 0, 0);
    };

    const int EW_BLK = 256;
    const int NB_ACT = (int)(ACT / EW_BLK);               // 32768

    // 1) mu-law
    mulaw_kernel<<<(BATCH * T_LEN) / EW_BLK, EW_BLK, 0, stream>>>(
        x_raw, xin, 1.0f / logf(257.0f));

    // 2) stacks
    for (int s = 0; s < 2; ++s) {
        // causal conv chain 1 -> 64 -> 128 -> 256
        conv_in_kernel<<<(BATCH * 64 * T_LEN) / EW_BLK, EW_BLK, 0, stream>>>(
            xin, sconv[s][0].W, sconv[s][0].b, hbuf);
        convL(hbuf, pk_sconv[s][0], sconv[s][1].b, abuf, 64, 128, 1, 2, 0, 0);
        convL(abuf, pk_sconv[s][1], sconv[s][2].b, hbuf, 128, 256, 1, 2, 0, 0);

        for (int blk = 0; blk < 2; ++blk) {
            dstack(hbuf, fprm[s][blk], pk_f[s][blk], zf);
            dstack(hbuf, gprm[s][blk], pk_g[s][blk], zg);
            gate_kernel<<<NB_ACT, EW_BLK, 0, stream>>>(zf, zg, abuf);
            // scale conv (k=1) + in-place residual into h
            convL(abuf, pk_sc[s][blk], scprm[s][blk].b, hbuf, 256, 256, 1, 1, 0, 1);
            skip_add_kernel<<<NB_ACT, EW_BLK, 0, stream>>>(
                skip, hbuf, (s == 0 && blk == 0) ? 1 : 0);
        }
    }

    // 3) head: relu(skip) -> conv_one + relu -> conv_two -> softmax
    relu_f32_to_f16_kernel<<<NB_ACT, EW_BLK, 0, stream>>>(skip, abuf);
    convL(abuf, pk_c1, c1.b, bbuf, 256, 256, 1, 1, /*relu*/1, 0);
    convL(bbuf, pk_c2, c2.b, zf,   256, 256, 1, 1, 0, 0);
    softmax_kernel<<<(BATCH * T_LEN * 32) / EW_BLK, EW_BLK, 0, stream>>>(
        zf, (float*)d_out);
}
